// GruKRegion_15341623181727
// MI455X (gfx1250) — compile-verified, matrix operand order checked by reading, not run on hardware
//
#include <hip/hip_runtime.h>
#include <hip/hip_bf16.h>

// ---------------------------------------------------------------------------
// GRU-K-Region on MI455X (gfx1250, wave32, WMMA)
//
// Pipeline per layer l:
//   gi[dir][n*T+t][0:384] = x[n*T+t] @ Wih[l,dir]^T + bih[l,dir]   (parallel WMMA GEMM)
//   serial scan:  gh = h @ Whh^T  (WMMA, Whh resident in LDS)
//                 r = sig(ir+hr+bhr); z = sig(iz+hz+bhz); n = tanh(in + r*(hn+bhn))
//                 h = (1-z)*n + z*h ;  x_out[n*T+t][dir*128 + j] = h
// Head: h1 = leaky(last @ W1^T + b1) via WMMA; out = h1 @ W2^T + b2 scalar.
// ---------------------------------------------------------------------------

typedef __attribute__((ext_vector_type(16))) _Float16 v16h;
typedef __attribute__((ext_vector_type(8)))  _Float16 v8h;
typedef __attribute__((ext_vector_type(8)))  float    v8f;

#define NSEQ   400
#define TSTEPS 200
#define NT     (NSEQ * TSTEPS)        // 80000
#define HH     128
#define G3     384                     // 3*H
#define IN2H   256                     // 2*H

// A/B fragment for v_wmma_f32_16x16x32_f16, per ISA 7.12.2:
// lane half 0 (lanes 0-15) holds K = {kBase+0..7, kBase+16..23}
// lane half 1 (lanes16-31) holds K = {kBase+8..15, kBase+24..31}
// => two contiguous 8 x f16 (16 B) chunks from the lane's row.
__device__ __forceinline__ v16h frag_row16(const _Float16* row, int kBase, int khalf) {
    union { v16h v; v8h h[2]; } u;
    u.h[0] = *(const v8h*)(row + kBase + khalf);
    u.h[1] = *(const v8h*)(row + kBase + 16 + khalf);
    return u.v;
}

__device__ __forceinline__ v8f wmma_f16(v16h a, v16h b, v8f c) {
    return __builtin_amdgcn_wmma_f32_16x16x32_f16(false, a, false, b, (short)0, c,
                                                  false, false);
}

__device__ __forceinline__ float sigmoidf(float x) {
    return 1.0f / (1.0f + __expf(-x));
}

// ---------------------------------------------------------------------------
// f32 -> f16 weight conversion (once per launch; deterministic)
// ---------------------------------------------------------------------------
__global__ void __launch_bounds__(256)
cvt_f32_f16_kernel(const float* __restrict__ src, _Float16* __restrict__ dst, int n) {
    int i = blockIdx.x * 256 + threadIdx.x;
    if (i < n) dst[i] = (_Float16)src[i];
}

// ---------------------------------------------------------------------------
// Layer 0 input projection: input size is 1 -> pure elementwise
// gi[dir][nt][g] = raw[nt] * Wih0[dir][g] + bih[l0,dir][g]
// ---------------------------------------------------------------------------
__global__ void __launch_bounds__(256)
gi_layer0_kernel(const float* __restrict__ raw,   // [NT]
                 const float* __restrict__ wih0,  // [2][384]
                 const float* __restrict__ bih0,  // [2][384] (layer 0 slice)
                 float* __restrict__ gibuf)       // [2][NT][384]
{
    size_t idx = (size_t)blockIdx.x * 256 + threadIdx.x;
    const size_t total = 2ull * NT * G3;
    if (idx >= total) return;
    int    g   = (int)(idx % G3);
    size_t nt  = (idx / G3) % NT;
    int    dir = (int)(idx / ((size_t)G3 * NT));
    gibuf[idx] = raw[nt] * wih0[dir * G3 + g] + bih0[dir * G3 + g];
}

// ---------------------------------------------------------------------------
// Layers 1..3 input projection: (80000 x 256) @ (256 x 384) per direction.
// Grid: (rowTile 0..4999, colGroup 0..2, dir 0..1), 8 waves/block,
// wave w owns colTile = colGroup*8 + w. One 16x16 C tile, K=256 -> 8 WMMA.
// ---------------------------------------------------------------------------
__global__ void __launch_bounds__(256)
gi_gemm_kernel(const _Float16* __restrict__ xin,   // [NT][256] f16
               const _Float16* __restrict__ wih,   // [2][384][256] f16 (layer slice)
               const float*    __restrict__ bih,   // [2][384] (layer slice)
               float*          __restrict__ gibuf) // [2][NT][384]
{
    const int tid   = threadIdx.x;
    const int lane  = tid & 31;
    const int wave  = tid >> 5;
    const int rowT  = blockIdx.x;
    const int colT  = blockIdx.y * 8 + wave;
    const int dir   = blockIdx.z;
    const int khalf = (lane >> 4) * 8;
    const int l15   = lane & 15;

    const _Float16* arow = xin + (size_t)(rowT * 16 + l15) * IN2H;
    const _Float16* brow = wih + ((size_t)dir * G3 + colT * 16 + l15) * IN2H;

    v8f c = {};
#pragma unroll
    for (int kb = 0; kb < IN2H; kb += 32) {
        v16h a = frag_row16(arow, kb, khalf);
        v16h b = frag_row16(brow, kb, khalf);
        c = wmma_f16(a, b, c);
    }

    const int   col   = colT * 16 + l15;
    const float bb    = bih[dir * G3 + col];
    const int   rbase = (lane >> 4) * 8;
    float* out = gibuf + ((size_t)dir * NT + (size_t)rowT * 16) * G3;
#pragma unroll
    for (int r = 0; r < 8; ++r)
        out[(size_t)(rbase + r) * G3 + col] = c[r] + bb;
}

// ---------------------------------------------------------------------------
// Recurrent scan, one layer. Grid (25 seq-tiles, 2 dirs), 256 threads.
// Whh (384x128 f16 = 96KB) resident in LDS for all 200 steps.
// h kept in LDS as f32 (master) + f16 (WMMA A feed).
// Per step per wave: 3 col tiles x 4 K-steps = 12 WMMA.
// ---------------------------------------------------------------------------
__global__ void __launch_bounds__(256)
gru_layer_kernel(const float*    __restrict__ gi,    // [2][NT][384]
                 const _Float16* __restrict__ whh16, // [2][384][128] f16 (layer slice)
                 const float*    __restrict__ bhh,   // [2][384] (layer slice)
                 _Float16*       __restrict__ xout)  // [NT][256] f16
{
    extern __shared__ char smem[];
    _Float16* sWhh = (_Float16*)smem;                              // 384*128 f16 = 96 KB
    _Float16* sH16 = (_Float16*)(smem + 98304);                    // 16*128 f16  =  4 KB
    float*    sH32 = (float*)   (smem + 98304 + 4096);             // 16*128 f32  =  8 KB
    float*    sGh  = (float*)   (smem + 98304 + 4096 + 8192);      // 16*384 f32  = 24 KB

    const int tid   = threadIdx.x;
    const int lane  = tid & 31;
    const int wave  = tid >> 5;
    const int n0    = blockIdx.x * 16;
    const int dir   = blockIdx.y;
    const int khalf = (lane >> 4) * 8;
    const int l15   = lane & 15;

    // stage Whh for this direction into LDS (reused 200x)
    const _Float16* wsrc = whh16 + (size_t)dir * (G3 * HH);
    for (int i = tid; i < G3 * HH; i += 256) sWhh[i] = wsrc[i];
    for (int i = tid; i < 16 * HH; i += 256) {
        sH32[i] = 0.0f;
        sH16[i] = (_Float16)0.0f;
    }
    __syncthreads();

    const float*  bh   = bhh + (size_t)dir * G3;
    const size_t  gdir = (size_t)dir * NT;

    for (int s = 0; s < TSTEPS; ++s) {
        const int t = dir ? (TSTEPS - 1 - s) : s;

        // prefetch next step's gi rows into cache while we do the WMMAs
        if (s + 1 < TSTEPS && tid < 16) {
            const int tn = dir ? (t - 1) : (t + 1);
            __builtin_prefetch(gi + (gdir + (size_t)(n0 + tid) * TSTEPS + tn) * G3, 0, 1);
        }

        // gh = h(16x128) @ Whh^T(128x384)
        const _Float16* ha = sH16 + (size_t)l15 * HH;
        v16h a0 = frag_row16(ha, 0,  khalf);
        v16h a1 = frag_row16(ha, 32, khalf);
        v16h a2 = frag_row16(ha, 64, khalf);
        v16h a3 = frag_row16(ha, 96, khalf);
#pragma unroll
        for (int i = 0; i < 3; ++i) {
            const int ct = wave + i * 8;               // column tile 0..23
            const _Float16* brow = sWhh + (size_t)(ct * 16 + l15) * HH;
            v8f c = {};
            c = wmma_f16(a0, frag_row16(brow, 0,  khalf), c);
            c = wmma_f16(a1, frag_row16(brow, 32, khalf), c);
            c = wmma_f16(a2, frag_row16(brow, 64, khalf), c);
            c = wmma_f16(a3, frag_row16(brow, 96, khalf), c);
            const int rbase = (lane >> 4) * 8;
#pragma unroll
            for (int r = 0; r < 8; ++r)
                sGh[(rbase + r) * G3 + ct * 16 + l15] = c[r];
        }
        __syncthreads();

        // gate math + h update + output store (256 threads x 8 elements)
        for (int e = tid; e < 16 * HH; e += 256) {
            const int m = e >> 7;
            const int j = e & (HH - 1);
            const size_t nt = (size_t)(n0 + m) * TSTEPS + t;
            const float* gir = gi + (gdir + nt) * G3;

            float ir  = gir[j];
            float iz  = gir[HH + j];
            float inn = gir[2 * HH + j];
            float hr  = sGh[m * G3 + j]          + bh[j];
            float hz  = sGh[m * G3 + HH + j]     + bh[HH + j];
            float hn  = sGh[m * G3 + 2 * HH + j] + bh[2 * HH + j];

            float r = sigmoidf(ir + hr);
            float z = sigmoidf(iz + hz);
            float n = tanhf(inn + r * hn);
            float h = (1.0f - z) * n + z * sH32[m * HH + j];

            sH32[m * HH + j] = h;
            sH16[m * HH + j] = (_Float16)h;
            xout[nt * IN2H + dir * HH + j] = (_Float16)h;
        }
        __syncthreads();
    }
}

// ---------------------------------------------------------------------------
// Head: last = x[:, T-1, 0:256]; h1 = leaky(last @ W1^T + b1) via WMMA;
// out = h1 @ W2^T + b2 via scalar dot. One 16-row tile per block.
// ---------------------------------------------------------------------------
__global__ void __launch_bounds__(256)
head_kernel(const _Float16* __restrict__ xin,  // [NT][256] f16
            const _Float16* __restrict__ w116, // [128][256] f16
            const float*    __restrict__ b1,   // [128]
            const float*    __restrict__ w2,   // [8][128]
            const float*    __restrict__ b2,   // [8]
            float*          __restrict__ out)  // [400][8]
{
    __shared__ float sH1[16 * HH];              // 8 KB
    const int tid   = threadIdx.x;
    const int lane  = tid & 31;
    const int wave  = tid >> 5;                 // colTile = wave (8 tiles x 16 = 128 cols)
    const int n0    = blockIdx.x * 16;
    const int khalf = (lane >> 4) * 8;
    const int l15   = lane & 15;

    const _Float16* arow = xin + ((size_t)(n0 + l15) * TSTEPS + (TSTEPS - 1)) * IN2H;
    const _Float16* brow = w116 + (size_t)(wave * 16 + l15) * IN2H;

    v8f c = {};
#pragma unroll
    for (int kb = 0; kb < IN2H; kb += 32) {
        v16h a = frag_row16(arow, kb, khalf);
        v16h b = frag_row16(brow, kb, khalf);
        c = wmma_f16(a, b, c);
    }
    const int col   = wave * 16 + l15;
    const float bb  = b1[col];
    const int rbase = (lane >> 4) * 8;
#pragma unroll
    for (int r = 0; r < 8; ++r) {
        float v = c[r] + bb;
        sH1[(rbase + r) * HH + col] = (v > 0.0f) ? v : 0.2f * v;   // LeakyReLU(0.2)
    }
    __syncthreads();

    if (tid < 128) {
        const int m = tid >> 3;
        const int o = tid & 7;
        float acc = b2[o];
#pragma unroll 8
        for (int k = 0; k < HH; ++k) acc += sH1[m * HH + k] * w2[o * HH + k];
        out[(size_t)(n0 + m) * 8 + o] = acc;
    }
}

// ---------------------------------------------------------------------------
// Host launch
// ---------------------------------------------------------------------------
extern "C" void kernel_launch(void* const* d_in, const int* in_sizes, int n_in,
                              void* d_out, int out_size, void* d_ws, size_t ws_size,
                              hipStream_t stream) {
    const float* raw  = (const float*)d_in[0];  // (4,100,200) == (400,200)
    const float* Wih0 = (const float*)d_in[1];  // (2,384,1)
    const float* Wih  = (const float*)d_in[2];  // (3,2,384,256)
    const float* Whh  = (const float*)d_in[3];  // (4,2,384,128)
    const float* bih  = (const float*)d_in[4];  // (4,2,384)
    const float* bhh  = (const float*)d_in[5];  // (4,2,384)
    const float* W1   = (const float*)d_in[6];  // (128,256)
    const float* b1   = (const float*)d_in[7];  // (128)
    const float* W2   = (const float*)d_in[8];  // (8,128)
    const float* b2   = (const float*)d_in[9];  // (8)
    float* out = (float*)d_out;                 // (4,100,8)

    // workspace layout
    char* ws = (char*)d_ws;
    _Float16* xbuf  = (_Float16*)ws;                               // NT*256 f16   = 40,960,000 B
    float*    gibuf = (float*)(ws + 40960000);                     // 2*NT*384 f32 = 245,760,000 B
    _Float16* wih16 = (_Float16*)(ws + 40960000 + 98304000);       // 3*2*384*256
    _Float16* whh16 = wih16 + 3 * 2 * G3 * IN2H;                   // 4*2*384*128
    _Float16* w116  = whh16 + 4 * 2 * G3 * HH;                     // 128*256

    const int nWih = 3 * 2 * G3 * IN2H;   // 589824
    const int nWhh = 4 * 2 * G3 * HH;     // 393216
    const int nW1  = HH * IN2H;           // 32768
    cvt_f32_f16_kernel<<<(nWih + 255) / 256, 256, 0, stream>>>(Wih, wih16, nWih);
    cvt_f32_f16_kernel<<<(nWhh + 255) / 256, 256, 0, stream>>>(Whh, whh16, nWhh);
    cvt_f32_f16_kernel<<<(nW1  + 255) / 256, 256, 0, stream>>>(W1,  w116,  nW1);

    const size_t gruSmem = 98304 + 4096 + 8192 + 24576;            // 135,168 B < 320 KB/WGP

    for (int l = 0; l < 4; ++l) {
        if (l == 0) {
            const size_t tot = 2ull * NT * G3;
            gi_layer0_kernel<<<(int)((tot + 255) / 256), 256, 0, stream>>>(
                raw, Wih0, bih /* layer 0 slice */, gibuf);
        } else {
            gi_gemm_kernel<<<dim3(NSEQ * TSTEPS / 16, 3, 2), 256, 0, stream>>>(
                xbuf,
                wih16 + (size_t)(l - 1) * 2 * G3 * IN2H,
                bih + (size_t)l * 2 * G3,
                gibuf);
        }
        gru_layer_kernel<<<dim3(NSEQ / 16, 2), 256, gruSmem, stream>>>(
            gibuf,
            whh16 + (size_t)l * 2 * G3 * HH,
            bhh + (size_t)l * 2 * G3,
            xbuf);
    }

    head_kernel<<<NSEQ / 16, 256, 0, stream>>>(xbuf, w116, b1, W2, b2, out);
}
